// Temporal_voting_FC1_action_89833535963828
// MI455X (gfx1250) — compile-verified, matrix operand order checked by reading, not run on hardware
//
#include <hip/hip_runtime.h>

typedef __bf16 bf16_t;
typedef __attribute__((ext_vector_type(16))) __bf16 v16bf;
typedef __attribute__((ext_vector_type(8)))  float  v8f;

#define T_TOKENS 262144
#define C_DIM    1152
#define K_CLS    285
#define K_PAD    288
#define N_TILES  18          // 288 / 16
#define K_STEPS  36          // 1152 / 32
#define WBF_BYTES (K_PAD * C_DIM * 2)   // bf16 W copy in d_ws

// ---------------------------------------------------------------------------
// Kernel 0: convert W [285x1152] f32 -> bf16 padded to [288][1152], zero hist
// ---------------------------------------------------------------------------
__global__ void prep_kernel(const float* __restrict__ W,
                            bf16_t* __restrict__ Wbf,
                            int* __restrict__ ghist) {
    int idx = blockIdx.x * blockDim.x + threadIdx.x;
    if (idx < K_PAD * C_DIM) {
        int k = idx / C_DIM;
        int c = idx - k * C_DIM;
        float v = (k < K_CLS) ? W[k * C_DIM + c] : 0.0f;
        Wbf[idx] = (bf16_t)v;
    }
    if (idx < K_PAD) ghist[idx] = 0;
}

// ---------------------------------------------------------------------------
// Kernel 1: bf16 WMMA GEMM (x @ W^T + b), per-row argmax, vote histogram.
// 256 threads = 8 waves; each wave: 16 token rows x 288 classes.
// ---------------------------------------------------------------------------
__launch_bounds__(256, 1)
__global__ void gemm_vote_kernel(const float* __restrict__ x,
                                 const bf16_t* __restrict__ Wbf,
                                 const float* __restrict__ bias,
                                 int* __restrict__ ghist) {
    __shared__ float s_bias[K_PAD];
    __shared__ int   s_hist[K_PAD];
    __shared__ float s_stage[8][256];   // per-wave 16x16 D-tile staging

    const int tid = threadIdx.x;
    for (int i = tid; i < K_PAD; i += 256) {
        s_bias[i] = (i < K_CLS) ? bias[i] : -3.0e38f;
        s_hist[i] = 0;
    }
    __syncthreads();

    const int wave = tid >> 5;
    const int lane = tid & 31;
    const int h    = lane >> 4;   // K-half selector (ISA 16-bit A/B layout)
    const int n    = lane & 15;   // row (A) / column (B) within tile

    const long trow = (long)blockIdx.x * 128 + (long)wave * 16 + n;
    const float* __restrict__ xrow = x + trow * C_DIM;

    v8f acc[N_TILES];
    const v8f vzero = {0.f, 0.f, 0.f, 0.f, 0.f, 0.f, 0.f, 0.f};
#pragma unroll
    for (int i = 0; i < N_TILES; ++i) acc[i] = vzero;

    for (int ks = 0; ks < K_STEPS; ++ks) {
        const int c0 = ks * 32 + h * 8;

        // ---- A: 16 fp32 of this token row -> packed bf16 (ISA A layout) ----
        float4 f0 = *(const float4*)(xrow + c0);
        float4 f1 = *(const float4*)(xrow + c0 + 4);
        float4 f2 = *(const float4*)(xrow + c0 + 16);
        float4 f3 = *(const float4*)(xrow + c0 + 20);
        v16bf a;
        a[0]  = (bf16_t)f0.x; a[1]  = (bf16_t)f0.y; a[2]  = (bf16_t)f0.z; a[3]  = (bf16_t)f0.w;
        a[4]  = (bf16_t)f1.x; a[5]  = (bf16_t)f1.y; a[6]  = (bf16_t)f1.z; a[7]  = (bf16_t)f1.w;
        a[8]  = (bf16_t)f2.x; a[9]  = (bf16_t)f2.y; a[10] = (bf16_t)f2.z; a[11] = (bf16_t)f2.w;
        a[12] = (bf16_t)f3.x; a[13] = (bf16_t)f3.y; a[14] = (bf16_t)f3.z; a[15] = (bf16_t)f3.w;

        // ---- B: pre-converted bf16 W rows (= B columns), 2x16B per tile ----
#pragma unroll
        for (int nt = 0; nt < N_TILES; ++nt) {
            const bf16_t* wrow = Wbf + (size_t)(nt * 16 + n) * C_DIM + c0;
            union { v16bf v; uint4 u[2]; } bm;
            bm.u[0] = *(const uint4*)(wrow);
            bm.u[1] = *(const uint4*)(wrow + 16);
            acc[nt] = __builtin_amdgcn_wmma_f32_16x16x32_bf16(
                false, a, false, bm.v, (short)0, acc[nt], false, false);
        }
    }

    // ---- per-row argmax via LDS staging of each 16x16 D tile ----
    float best  = -3.0e38f;
    int   bestk = 0;
#pragma unroll 1
    for (int nt = 0; nt < N_TILES; ++nt) {
#pragma unroll
        for (int j = 0; j < 8; ++j)
            s_stage[wave][(j + 8 * h) * 16 + n] = acc[nt][j];
        __syncthreads();
        if (lane < 16) {
#pragma unroll
            for (int jj = 0; jj < 16; ++jj) {
                int kidx = nt * 16 + jj;
                float v = s_stage[wave][lane * 16 + jj] + s_bias[kidx];
                if (v > best) { best = v; bestk = kidx; }
            }
        }
        __syncthreads();
    }

    if (lane < 16) atomicAdd(&s_hist[bestk], 1);
    __syncthreads();

    for (int i = tid; i < K_CLS; i += 256) {
        int c = s_hist[i];
        if (c) atomicAdd(&ghist[i], c);
    }
}

// ---------------------------------------------------------------------------
// Kernel 2: histogram argmax (ties -> lowest index) and one-hot write.
// ---------------------------------------------------------------------------
__global__ void finalize_kernel(const int* __restrict__ ghist,
                                float* __restrict__ out) {
    __shared__ int sval[512];
    __shared__ int sidx[512];
    const int tid = threadIdx.x;
    sval[tid] = (tid < K_CLS) ? ghist[tid] : -1;
    sidx[tid] = tid;
    __syncthreads();
    for (int s = 256; s > 0; s >>= 1) {
        if (tid < s) {
            int v2 = sval[tid + s], i2 = sidx[tid + s];
            if (v2 > sval[tid] || (v2 == sval[tid] && i2 < sidx[tid])) {
                sval[tid] = v2; sidx[tid] = i2;
            }
        }
        __syncthreads();
    }
    if (tid < K_CLS) out[tid] = (tid == sidx[0]) ? 1.0f : 0.0f;
}

// ---------------------------------------------------------------------------
extern "C" void kernel_launch(void* const* d_in, const int* in_sizes, int n_in,
                              void* d_out, int out_size, void* d_ws, size_t ws_size,
                              hipStream_t stream) {
    const float* x = (const float*)d_in[0];   // [1, 262144, 1152] f32
    const float* W = (const float*)d_in[1];   // [285, 1152] f32
    const float* b = (const float*)d_in[2];   // [285] f32

    bf16_t* Wbf   = (bf16_t*)d_ws;
    int*    ghist = (int*)((char*)d_ws + WBF_BYTES);
    float*  out   = (float*)d_out;            // [1, 285] f32 one-hot

    prep_kernel<<<(K_PAD * C_DIM + 255) / 256, 256, 0, stream>>>(W, Wbf, ghist);
    gemm_vote_kernel<<<T_TOKENS / 128, 256, 0, stream>>>(x, Wbf, b, ghist);
    finalize_kernel<<<1, 512, 0, stream>>>(ghist, out);
}
